// ODEFunc_36232344109290
// MI455X (gfx1250) — matrix operand order repaired and checked by reading, not compiled, not run on hardware
//
#include <hip/hip_runtime.h>
#include <hip/hip_bf16.h>
#include <math.h>

// ---------------------------------------------------------------------------
// DCRNN ODE func for MI455X (gfx1250).
//   CSR built per-launch (count + LDS scan + scatter) -> atomic-free pull SpMM.
//   Phase 1: diffusion of x (Fin=64) -> X1 (80000 x 320) bf16
//            theta = sigmoid(X1 @ W_lat + b_lat)      [WMMA bf16, f32 acc]
//            c1    = tanh  (X1 @ W_units + b_units)   [WMMA bf16, f32 acc]
//   Phase 2: diffusion of c1 (Fin=128) -> X2 (80000 x 640) bf16
//            out = -theta * tanh(X2 @ W_final + b_lat) [fused epilogue]
// GEMM: weights staged in LDS (ds_load_b128 B fragments); one wave owns a
// full 16-row strip so the A fragment is register-reused across NT WMMAs.
// ---------------------------------------------------------------------------

typedef __attribute__((ext_vector_type(16))) __bf16 v16bf;
typedef __attribute__((ext_vector_type(8)))  __bf16 v8bf;
typedef __attribute__((ext_vector_type(8)))  float  v8f;

#define NODES 10000
#define FLAT  64
#define UNITS 128
#define BATCH 8
#define MSUP  5

__device__ __forceinline__ unsigned short f32_to_bf16_rne(float f) {
  unsigned int u = __float_as_uint(f);
  u += 0x7FFFu + ((u >> 16) & 1u);
  return (unsigned short)(u >> 16);
}

// ------------------------- CSR construction -------------------------------
__global__ void count_edges(const int* __restrict__ dst, int* __restrict__ cnt, int ne) {
  int e = blockIdx.x * blockDim.x + threadIdx.x;
  if (e < ne) atomicAdd(&cnt[dst[e]], 1);
}

// Single-block exclusive scan over n (<= a few 10k) counts -> ptr[0..n], cursor copy.
__global__ void exclusive_scan(const int* __restrict__ counts, int* __restrict__ ptr,
                               int* __restrict__ cursor, int n) {
  __shared__ int sdata[1024];
  __shared__ int carry;
  if (threadIdx.x == 0) carry = 0;
  __syncthreads();
  for (int base = 0; base < n; base += 1024) {
    int i = base + (int)threadIdx.x;
    int v = (i < n) ? counts[i] : 0;
    sdata[threadIdx.x] = v;
    __syncthreads();
    for (int off = 1; off < 1024; off <<= 1) {
      int t = (threadIdx.x >= (unsigned)off) ? sdata[threadIdx.x - off] : 0;
      __syncthreads();
      sdata[threadIdx.x] += t;
      __syncthreads();
    }
    int excl = sdata[threadIdx.x] - v + carry;
    if (i < n) { ptr[i] = excl; cursor[i] = excl; }
    int blocktotal = sdata[1023];
    __syncthreads();
    if (threadIdx.x == 0) carry += blocktotal;
    __syncthreads();
  }
  if (threadIdx.x == 0) ptr[n] = carry;
}

__global__ void scatter_edges(const int* __restrict__ dst, const int* __restrict__ src,
                              const float* __restrict__ w, int* __restrict__ cursor,
                              int* __restrict__ srcs, float* __restrict__ wsrt, int ne) {
  int e = blockIdx.x * blockDim.x + threadIdx.x;
  if (e >= ne) return;
  int pos = atomicAdd(&cursor[dst[e]], 1);
  srcs[pos] = src[e];
  wsrt[pos] = w[e];
}

// ---------------- pull-style SpMM (no atomics) ----------------------------
// xout[nd] = sum_j w[j] * xin[srcs[j]]  over CSR row nd; one thread per
// (node, float4 chunk of the Fin*B row).
__global__ void spmm_csr(const float* __restrict__ xin, float* __restrict__ xout,
                         const int* __restrict__ ptr, const int* __restrict__ srcs,
                         const float* __restrict__ wsrt, int finb) {
  int chunks = finb >> 2;
  long long tid = (long long)blockIdx.x * blockDim.x + threadIdx.x;
  long long total = (long long)NODES * chunks;
  if (tid >= total) return;
  int nd = (int)(tid / chunks);
  int c  = (int)(tid % chunks);
  int beg = ptr[nd], end = ptr[nd + 1];
  float ax = 0.f, ay = 0.f, az = 0.f, aw = 0.f;
  for (int j = beg; j < end; ++j) {
    float wv = wsrt[j];
    const float4 v = ((const float4*)(xin + (long long)srcs[j] * finb))[c];
    ax += wv * v.x; ay += wv * v.y; az += wv * v.z; aw += wv * v.w;
  }
  float4 r; r.x = ax; r.y = ay; r.z = az; r.w = aw;
  ((float4*)(xout + (long long)nd * finb))[c] = r;
}

// (B, N, Fin) -> (N, Fin, B)
__global__ void transpose_bnf_to_nfb(const float* __restrict__ in,
                                     float* __restrict__ out, int fin) {
  long long tid = (long long)blockIdx.x * blockDim.x + threadIdx.x;
  long long total = (long long)BATCH * NODES * fin;
  if (tid >= total) return;
  int b = (int)(tid % BATCH);
  long long t = tid / BATCH;
  int f = (int)(t % fin);
  int n = (int)(t / fin);
  out[tid] = in[((long long)b * NODES + n) * fin + f];
}

// Pack + fused Chebyshev: one thread per (row=b*N+n, f) writes 5 adjacent cols.
//   Xb[row*Fin*5 + f*5 + m], m in {x0, S1x0, 2*S1t2-x0, S2x0, 2*S2t4-x0}
__global__ void pack_bf16(const float* __restrict__ x0, const float* __restrict__ x1,
                          const float* __restrict__ t2, const float* __restrict__ x3,
                          const float* __restrict__ t4,
                          unsigned short* __restrict__ Xb, int fin) {
  long long tid = (long long)blockIdx.x * blockDim.x + threadIdx.x;
  long long total = (long long)BATCH * NODES * fin;
  if (tid >= total) return;
  int f = (int)(tid % fin);
  long long row = tid / fin;              // b*NODES + n
  int n = (int)(row % NODES);
  int b = (int)(row / NODES);
  long long idx = ((long long)n * fin + f) * BATCH + b;
  float v0 = x0[idx];
  unsigned short* o = Xb + tid * MSUP;
  o[0] = f32_to_bf16_rne(v0);
  o[1] = f32_to_bf16_rne(x1[idx]);
  o[2] = f32_to_bf16_rne(2.0f * t2[idx] - v0);
  o[3] = f32_to_bf16_rne(x3[idx]);
  o[4] = f32_to_bf16_rne(2.0f * t4[idx] - v0);
}

// Wt[n*ktot + k] = bf16(W[k*nout + n])
__global__ void wtrans_bf16(const float* __restrict__ W,
                            unsigned short* __restrict__ Wt, int ktot, int nout) {
  int tid = blockIdx.x * blockDim.x + threadIdx.x;
  if (tid >= ktot * nout) return;
  int k = tid % ktot;
  int n = tid / ktot;
  Wt[tid] = f32_to_bf16_rne(W[(long long)k * nout + n]);
}

// C(rows x NT*16) = act( A(rows x KT, bf16) @ Bt^T + bias )
// Bt (NT*16 x KT bf16) staged in LDS; one wave owns a full 16-row strip.
// ACT: 1=sigmoid, 2=tanh, 3 = -theta * tanh(.)
template <int NT, int ACT, int KT>
__global__ void wmma_gemm(const unsigned short* __restrict__ A,
                          const unsigned short* __restrict__ Bt,
                          const float* __restrict__ bias,
                          const float* __restrict__ theta,
                          float* __restrict__ out) {
  extern __shared__ unsigned short Bs[];
  const int nout = NT * 16;
  const int nelem = KT * nout;
  for (int i = threadIdx.x * 8; i < nelem; i += blockDim.x * 8)
    *(uint4*)(Bs + i) = *(const uint4*)(Bt + i);
  __syncthreads();

  int tm   = blockIdx.x * (blockDim.x >> 5) + (threadIdx.x >> 5);
  int lane = threadIdx.x & 31;
  int half = lane >> 4;   // 0: lanes 0-15, 1: lanes 16-31
  int l16  = lane & 15;

  // A frag (16x32 bf16): lane<16: row=l16, K 0..15;
  //                      lane>=16: K 16..31
  const __bf16* Arow  = (const __bf16*)A + (long long)(tm * 16 + l16) * KT + half * 16;
  // B frags (32x16 bf16) served from LDS: col=l16, K half by lane group
  const __bf16* Bbase = (const __bf16*)Bs + l16 * KT + half * 16;

  v8f acc[NT];
  const v8f vz = {};
#pragma unroll
  for (int t = 0; t < NT; ++t) acc[t] = vz;

  for (int k0 = 0; k0 < KT; k0 += 32) {
    v8bf a0 = *(const v8bf*)(Arow + k0);
    v8bf a1 = *(const v8bf*)(Arow + k0 + 8);
    __builtin_prefetch(Arow + k0 + 256, 0, 1);   // global_prefetch_b8
    v16bf av;
#pragma unroll
    for (int i = 0; i < 8; ++i) { av[i] = a0[i]; av[i + 8] = a1[i]; }
#pragma unroll
    for (int t = 0; t < NT; ++t) {
      v16bf bv = *(const v16bf*)(Bbase + t * 16 * KT + k0);   // ds_load
      acc[t] = __builtin_amdgcn_wmma_f32_16x16x32_bf16(
          /*neg_a=*/false, av, /*neg_b=*/false, bv,
          /*c_mod=*/(short)0, acc[t], /*reuse_a=*/false, /*reuse_b=*/false);
    }
  }

  // C/D layout: VGPR r -> M = r + 8*half ; N = l16 within each tile
#pragma unroll
  for (int t = 0; t < NT; ++t) {
    int n = t * 16 + l16;
    float bvv = bias[n];
#pragma unroll
    for (int r = 0; r < 8; ++r) {
      long long row = (long long)tm * 16 + r + half * 8;
      float v = acc[t][r] + bvv;
      if (ACT == 1)      v = 1.0f / (1.0f + __expf(-v));
      else if (ACT == 2) v = tanhf(v);
      else               v = -theta[row * nout + n] * tanhf(v);
      out[row * nout + n] = v;
    }
  }
}

static inline int gridFor(long long n) { return (int)((n + 255) / 256); }

extern "C" void kernel_launch(void* const* d_in, const int* in_sizes, int n_in,
                              void* d_out, int out_size, void* d_ws, size_t ws_size,
                              hipStream_t stream) {
  const float* y       = (const float*)d_in[0];
  const float* W_lat   = (const float*)d_in[1];
  const float* b_lat   = (const float*)d_in[2];
  const float* W_units = (const float*)d_in[3];
  const float* b_units = (const float*)d_in[4];
  const float* W_final = (const float*)d_in[5];
  const float* s1w     = (const float*)d_in[6];
  const float* s2w     = (const float*)d_in[7];
  const int*   erow    = (const int*)d_in[8];
  const int*   ecol    = (const int*)d_in[9];
  const int E = in_sizes[8];
  float* outp = (float*)d_out;
  char* ws = (char*)d_ws;

  // ---- workspace layout (bytes, lifetime-aliased; peak ~331 MB) ----
  const long long NFB = (long long)NODES * FLAT * BATCH;     //  5,120,000 f32
  const long long NUB = (long long)NODES * UNITS * BATCH;    // 10,240,000 f32
  float* theta = (float*)(ws + 0);                           // lives whole launch
  float* xt0   = (float*)(ws + 20480000);
  float* xs1   = (float*)(ws + 40960000);
  float* xs2   = xs1 + NFB;    // raw S1 @ xs1 (combine fused into pack)
  float* xs3   = xs2 + NFB;
  float* xs4   = xs3 + NFB;    // raw S2 @ xs3
  unsigned short* Xb1 = (unsigned short*)(ws + 122880000);   // 80000 x 320 bf16
  float* c1    = (float*)(ws + 174080000);                   // (B,N,U)
  // phase 2 reuses phase-1 regions (all dead by then):
  float* ct    = (float*)(ws + 20480000);                    // (N,U,B)
  float* cs1   = (float*)(ws + 61440000);
  float* cs2   = cs1 + NUB;
  float* cs3   = cs2 + NUB;
  float* cs4   = cs3 + NUB;
  unsigned short* Xb2   = (unsigned short*)(ws + 225280000); // 80000 x 640 bf16
  unsigned short* WlatT = (unsigned short*)(ws + 327680000); // 64  x 320
  unsigned short* WuniT = (unsigned short*)(ws + 327720960); // 128 x 320
  unsigned short* WfinT = (unsigned short*)(ws + 327802880); // 64  x 640
  // CSR region (persistent for whole launch)
  int*   ptr1 = (int*)  (ws + 327884800);                    // N+1 ints
  int*   ptr2 = (int*)  (ws + 327925760);
  int*   cur1 = (int*)  (ws + 327966720);
  int*   cur2 = (int*)  (ws + 328007680);
  int*   cnt1 = (int*)  (ws + 328048640);
  int*   cnt2 = (int*)  (ws + 328089600);
  int*   srcs1= (int*)  (ws + 328130560);                    // E ints
  int*   srcs2= (int*)  (ws + 328770560);
  float* wsr1 = (float*)(ws + 329410560);                    // E floats
  float* wsr2 = (float*)(ws + 330050560);

  const int T = 256;
  (void)n_in; (void)out_size; (void)ws_size;

  // ---- transpose weights to bf16 (Nout x K) for B-fragment LDS staging
  wtrans_bf16<<<gridFor(320 * 64),  T, 0, stream>>>(W_lat,   WlatT, 320, 64);
  wtrans_bf16<<<gridFor(320 * 128), T, 0, stream>>>(W_units, WuniT, 320, 128);
  wtrans_bf16<<<gridFor(640 * 64),  T, 0, stream>>>(W_final, WfinT, 640, 64);

  // ---- build CSR once per launch (reused by all 8 SpMMs)
  // support1: dst=cols, src=rows, w1 ; support2: dst=rows, src=cols, w2
  hipMemsetAsync(cnt1, 0, 2 * 40960, stream);
  count_edges<<<gridFor(E), T, 0, stream>>>(ecol, cnt1, E);
  count_edges<<<gridFor(E), T, 0, stream>>>(erow, cnt2, E);
  exclusive_scan<<<1, 1024, 0, stream>>>(cnt1, ptr1, cur1, NODES);
  exclusive_scan<<<1, 1024, 0, stream>>>(cnt2, ptr2, cur2, NODES);
  scatter_edges<<<gridFor(E), T, 0, stream>>>(ecol, erow, s1w, cur1, srcs1, wsr1, E);
  scatter_edges<<<gridFor(E), T, 0, stream>>>(erow, ecol, s2w, cur2, srcs2, wsr2, E);

  // ================= Phase 1: shared diffusion of x (Fin=64) ================
  transpose_bnf_to_nfb<<<gridFor(NFB), T, 0, stream>>>(y, xt0, FLAT);
  const int FB1 = FLAT * BATCH;                 // 512
  const long long g1 = (long long)NODES * (FB1 / 4);
  spmm_csr<<<gridFor(g1), T, 0, stream>>>(xt0, xs1, ptr1, srcs1, wsr1, FB1);
  spmm_csr<<<gridFor(g1), T, 0, stream>>>(xt0, xs3, ptr2, srcs2, wsr2, FB1);
  spmm_csr<<<gridFor(g1), T, 0, stream>>>(xs1, xs2, ptr1, srcs1, wsr1, FB1);
  spmm_csr<<<gridFor(g1), T, 0, stream>>>(xs3, xs4, ptr2, srcs2, wsr2, FB1);
  pack_bf16<<<gridFor(NFB), T, 0, stream>>>(xt0, xs1, xs2, xs3, xs4, Xb1, FLAT);

  // 5000 M-tiles, 8 waves/block -> 625 blocks; each wave does all N tiles
  wmma_gemm<4, 1, 320><<<625, 256, 320 * 64 * 2, stream>>>(Xb1, WlatT, b_lat,   nullptr, theta);
  wmma_gemm<8, 2, 320><<<625, 256, 320 * 128 * 2, stream>>>(Xb1, WuniT, b_units, nullptr, c1);

  // ================= Phase 2: diffusion of c1 (Fin=128) =====================
  transpose_bnf_to_nfb<<<gridFor(NUB), T, 0, stream>>>(c1, ct, UNITS);
  const int FB2 = UNITS * BATCH;                // 1024
  const long long g2 = (long long)NODES * (FB2 / 4);
  spmm_csr<<<gridFor(g2), T, 0, stream>>>(ct, cs1, ptr1, srcs1, wsr1, FB2);
  spmm_csr<<<gridFor(g2), T, 0, stream>>>(ct, cs3, ptr2, srcs2, wsr2, FB2);
  spmm_csr<<<gridFor(g2), T, 0, stream>>>(cs1, cs2, ptr1, srcs1, wsr1, FB2);
  spmm_csr<<<gridFor(g2), T, 0, stream>>>(cs3, cs4, ptr2, srcs2, wsr2, FB2);
  pack_bf16<<<gridFor(NUB), T, 0, stream>>>(ct, cs1, cs2, cs3, cs4, Xb2, UNITS);

  // out = -theta * tanh(X2 @ W_final + b_lat)
  wmma_gemm<4, 3, 640><<<625, 256, 640 * 64 * 2, stream>>>(Xb2, WfinT, b_lat, theta, outp);
}
